// EarthAttention3D_64235530879383
// MI455X (gfx1250) — compile-verified, matrix operand order checked by reading, not run on hardware
//
#include <hip/hip_runtime.h>
#include <hip/hip_bf16.h>

typedef __attribute__((ext_vector_type(16))) _Float16 v16h;
typedef __attribute__((ext_vector_type(8)))  _Float16 v8h;
typedef __attribute__((ext_vector_type(4)))  _Float16 v4h;
typedef __attribute__((ext_vector_type(8)))  float    v8f;
typedef __attribute__((ext_vector_type(4)))  unsigned int u32x4;
typedef __attribute__((ext_vector_type(8)))  int      i32x8;
typedef __attribute__((ext_vector_type(4)))  int      i32x4;

#define BDIM 96
#define NW   10
#define NTOK 144
#define CDIM 192
#define HEADS 6
#define DH   32

#if defined(__HIP_DEVICE_COMPILE__) && __has_builtin(__builtin_amdgcn_tensor_load_to_lds)
#define HAVE_TDM 1
#else
#define HAVE_TDM 0
#endif

// A/B fragment loader for v_wmma_f32_16x16x32_f16.
// A: 16x32 (MxK) row-major, lane l in [0,16): row M=l, k = {kb0..kb0+7, kb0+16..kb0+23}
//    lane l in [16,32): row M=l-16, k = {kb0+8..kb0+15, kb0+24..kb0+31}
// B: identical pattern with N in place of M (so B source must be stored N-major, i.e. transposed).
__device__ __forceinline__ v16h load_frag(const _Float16* __restrict__ base,
                                          int row0, int stride, int kb0, int lane) {
  int r  = row0 + (lane & 15);
  int kb = kb0 + ((lane >> 4) << 3);
  const _Float16* p = base + (size_t)r * stride + kb;
  v8h lo = *(const v8h*)(p);
  v8h hi = *(const v8h*)(p + 16);
  v16h o;
#pragma unroll
  for (int i = 0; i < 8; ++i) { o[i] = lo[i]; o[i + 8] = hi[i]; }
  return o;
}

#if HAVE_TDM
// TDM: 2D f16 tile load global->LDS with optional LDS row padding.
// D# per CDNA5 ISA ch.8: group0 = {count/lds_addr/global_addr/type}, group1 = dims/strides/pad.
// pad_interval: pad every 2^(v+1) DWORDs; pad_amount: (v+1) DWORDs inserted.
// This toolchain (clang-23 / therock-10.0) exposes the 6-arg builtin:
//   (u32x4 g0, i32x8 g1, i32x4 g2, i32x4 g3, i32x8 extra, i32 cpol)
__device__ __forceinline__ void tdm_load_2d_f16(
    unsigned int lds_byte_addr, const void* gptr,
    unsigned int tdim0, unsigned int tdim1, unsigned int tstride0,
    unsigned int tile0, unsigned int tile1,
    unsigned int pad_en, unsigned int pad_interval, unsigned int pad_amount) {
  unsigned long long ga = (unsigned long long)(uintptr_t)gptr;
  u32x4 g0;
  g0[0] = 1u;                                        // count=1 (valid), user mode
  g0[1] = lds_byte_addr;                             // lds_addr [63:32]
  g0[2] = (unsigned int)(ga & 0xffffffffu);          // global_addr [95:64]
  g0[3] = (unsigned int)((ga >> 32) & 0x01ffffffu)   // global_addr [120:96]
        | (2u << 30);                                // type=2 ("image") [127:126]
  i32x8 g1;
  g1[0] = (int)((1u << 16)                           // data_size=1 -> 2 bytes
        | (pad_en << 20) | (pad_interval << 22) | (pad_amount << 25));
  g1[1] = (int)(tdim0 << 16);                        // tensor_dim0[15:0] @ [63:48]
  g1[2] = (int)((tdim0 >> 16) | (tdim1 << 16));      // dim0 hi | dim1 lo
  g1[3] = (int)((tdim1 >> 16) | (tile0 << 16));      // dim1 hi | tile_dim0 @ [127:112]
  g1[4] = (int)(tile1 & 0xffffu);                    // tile_dim1 @ [143:128], tile_dim2=0
  g1[5] = (int)tstride0;                             // tensor_dim0_stride [207:160] lo32
  g1[6] = 0;
  g1[7] = 0;                                         // tensor_dim1_stride=0 (2D)
  i32x4 z4 = {0, 0, 0, 0};                           // groups 2/3 unused (2D tensor)
  i32x8 z8 = {0, 0, 0, 0, 0, 0, 0, 0};
  __builtin_amdgcn_tensor_load_to_lds(g0, g1, z4, z4, z8, 0);
}
#endif

// ---- prep: transpose + convert weights to f16, output-channel-major [Nout][K=192]
__global__ void k_prep_w(const float* __restrict__ w, _Float16* __restrict__ wT,
                         int rows /*K*/, int cols /*Nout*/) {
  int i = blockIdx.x * 256 + threadIdx.x;
  if (i >= rows * cols) return;
  int n = i / rows, k = i % rows;
  wT[(size_t)n * rows + k] = (_Float16)w[(size_t)k * cols + n];
}

// ---- prep: gather bias_table[position_index] -> dense [nW*H][N][N] fp32
__global__ void k_prep_bias(const float* __restrict__ table, const int* __restrict__ pidx,
                            float* __restrict__ biasG) {
  int i = blockIdx.x * 256 + threadIdx.x;
  const int total = NW * HEADS * NTOK * NTOK;
  if (i >= total) return;
  int j = i % NTOK;
  int r = (i / NTOK) % NTOK;
  int h = (i / (NTOK * NTOK)) % HEADS;
  int w = i / (NTOK * NTOK * HEADS);
  int p = pidx[r * NTOK + j];
  biasG[(size_t)((w * HEADS + h) * NTOK + r) * NTOK + j] =
      table[(size_t)p * (NW * HEADS) + w * HEADS + h];
}

// ---- QKV projection: one block per (b,w). x[144x192] -> Q,K (row-major, Q pre-scaled), V^T [Dh][N]
__global__ __launch_bounds__(256) void k_qkv(
    const float* __restrict__ x, const _Float16* __restrict__ wT,
    const float* __restrict__ bqkv,
    _Float16* __restrict__ Q, _Float16* __restrict__ Kt, _Float16* __restrict__ Vt) {
  __shared__ _Float16 As[NTOK * 200];
  int bw = blockIdx.x;
  const float* xp = x + (size_t)bw * NTOK * CDIM;
  int tid = threadIdx.x, lane = tid & 31, wave = tid >> 5;

  for (int idx = tid; idx < NTOK * CDIM / 4; idx += 256) {
    int e = idx * 4, row = e / CDIM, col = e % CDIM;
    float4 f = *(const float4*)(xp + e);
    v4h h4;
    h4[0] = (_Float16)f.x; h4[1] = (_Float16)f.y;
    h4[2] = (_Float16)f.z; h4[3] = (_Float16)f.w;
    *(v4h*)&As[row * 200 + col] = h4;
  }
  __syncthreads();

  const float scale = 0.17677669529663687f;  // 32^-0.5
  for (int t = wave; t < 9 * 36; t += 8) {
    int mt = t / 36, nt = t % 36;
    int m0 = mt * 16, n0 = nt * 16;
    int col = n0 + (lane & 15);
    float bv = bqkv[col];
    v8f acc;
#pragma unroll
    for (int r = 0; r < 8; ++r) acc[r] = bv;
#pragma unroll
    for (int kk = 0; kk < 6; ++kk) {
      v16h a  = load_frag(As, m0, 200, kk * 32, lane);
      v16h bb = load_frag(wT, n0, CDIM, kk * 32, lane);
      acc = __builtin_amdgcn_wmma_f32_16x16x32_f16(false, a, false, bb, (short)0, acc, false, false);
    }
    int which = n0 / CDIM, rem = n0 % CDIM, h = rem / DH, d0 = rem % DH;
    size_t head = (size_t)bw * HEADS + h;
    int rb = m0 + ((lane >> 4) << 3);
    if (which == 0) {
      _Float16* qp = Q + head * NTOK * DH;
      int c = d0 + (lane & 15);
#pragma unroll
      for (int r = 0; r < 8; ++r) qp[(rb + r) * DH + c] = (_Float16)(acc[r] * scale);
    } else if (which == 1) {
      _Float16* kp = Kt + head * NTOK * DH;
      int c = d0 + (lane & 15);
#pragma unroll
      for (int r = 0; r < 8; ++r) kp[(rb + r) * DH + c] = (_Float16)acc[r];
    } else {
      _Float16* vpp = Vt + head * DH * NTOK;
      int d = d0 + (lane & 15);
      v8h pk;
#pragma unroll
      for (int r = 0; r < 8; ++r) pk[r] = (_Float16)acc[r];
      *(v8h*)&vpp[(size_t)d * NTOK + rb] = pk;  // 8 consecutive rows -> one b128 store
    }
  }
}

// ---- Fused attention: one block per (b,w,h). S=QK^T (+bias+mask) -> softmax -> P·V
__global__ __launch_bounds__(256) void k_attn(
    const _Float16* __restrict__ Q, const _Float16* __restrict__ Km,
    const _Float16* __restrict__ Vt, const float* __restrict__ biasG,
    const float* __restrict__ mask, _Float16* __restrict__ AO) {
  extern __shared__ char smem[];
  _Float16* Qs  = (_Float16*)smem;                 // [144][40]
  _Float16* Ks  = Qs + 144 * 40;                   // [144][40]
  _Float16* Vts = Ks + 144 * 40;                   // [32][168] (cols 144..159 zero)
  float*    Ss  = (float*)(Vts + 32 * 168);        // [144][148] fp32 logits
  _Float16* Ps  = (_Float16*)(Ss + 144 * 148);     // [144][168] f16 unnormalized exp
  float*    rsum = (float*)(Ps + 144 * 168);       // [144] reciprocal row sums

  int bh = blockIdx.x;
  int b  = bh / (NW * HEADS);
  int wh = bh % (NW * HEADS);
  int h  = wh % HEADS;
  int tid = threadIdx.x, lane = tid & 31, wave = tid >> 5;

  const _Float16* qp = Q  + (size_t)bh * NTOK * DH;
  const _Float16* kp = Km + (size_t)bh * NTOK * DH;
  const _Float16* vp = Vt + (size_t)bh * DH * NTOK;

#if HAVE_TDM
  // TDM stages Q and K: 144 rows x 32 f16 (16 DWORDs) -> LDS rows padded +4 DWORDs
  // (stride 40 halves, conflict-free for the fragment loads). One descriptor per wave,
  // EXEC ignored, tracked by per-wave TENSORcnt.
  if (wave == 0) {
    tdm_load_2d_f16((unsigned int)(uintptr_t)(void*)Qs, qp,
                    /*tdim0=*/DH, /*tdim1=*/NTOK, /*tstride0=*/DH,
                    /*tile0=*/DH, /*tile1=*/NTOK,
                    /*pad_en=*/1u, /*pad_interval=*/3u /*16 DW*/, /*pad_amount=*/3u /*4 DW*/);
  } else if (wave == 1) {
    tdm_load_2d_f16((unsigned int)(uintptr_t)(void*)Ks, kp,
                    DH, NTOK, DH, DH, NTOK, 1u, 3u, 3u);
  }
#else
  for (int c = tid; c < NTOK * 4; c += 256) {      // 16B chunks of Q and K rows
    int row = c >> 2, co = (c & 3) * 8;
    *(v8h*)&Qs[row * 40 + co] = *(const v8h*)&qp[row * DH + co];
    *(v8h*)&Ks[row * 40 + co] = *(const v8h*)&kp[row * DH + co];
  }
#endif
  for (int c = tid; c < DH * 18; c += 256) {       // V^T rows (144 halves = 18 chunks)
    int row = c / 18, co = (c % 18) * 8;
    *(v8h*)&Vts[row * 168 + co] = *(const v8h*)&vp[row * NTOK + co];
  }
  if (tid < DH * 2) {                              // zero K-pad cols 144..159
    int row = tid >> 1, co = 144 + (tid & 1) * 8;
    v8h z = {};
    *(v8h*)&Vts[row * 168 + co] = z;
  }
#if HAVE_TDM
  if (wave < 2) __builtin_amdgcn_s_wait_tensorcnt(0);
#endif
  __syncthreads();

  // S = Q K^T : 81 tiles, exactly one K=32 WMMA each
  for (int t = wave; t < 81; t += 8) {
    int mt = t / 9, nt = t % 9;
    v16h a  = load_frag(Qs, mt * 16, 40, 0, lane);
    v16h bb = load_frag(Ks, nt * 16, 40, 0, lane);
    v8f acc = {};
    acc = __builtin_amdgcn_wmma_f32_16x16x32_f16(false, a, false, bb, (short)0, acc, false, false);
    int col = nt * 16 + (lane & 15);
    int rb  = mt * 16 + ((lane >> 4) << 3);
#pragma unroll
    for (int r = 0; r < 8; ++r) Ss[(rb + r) * 148 + col] = acc[r];
  }
  __syncthreads();

  // row softmax (unnormalized exp kept in f16, reciprocal sum applied post-GEMM)
  if (tid < NTOK) {
    const float* brow = biasG + (size_t)wh * NTOK * NTOK + (size_t)tid * NTOK;
    const float* mrow = mask  + (size_t)b  * NTOK * NTOK + (size_t)tid * NTOK;
    float* srow = &Ss[tid * 148];
    float mx = -3.0e38f;
    for (int j = 0; j < NTOK; ++j) {
      float s = srow[j] + brow[j] + mrow[j];
      srow[j] = s;
      mx = fmaxf(mx, s);
    }
    float sum = 0.f;
    _Float16* prow = &Ps[tid * 168];
    for (int j = 0; j < NTOK; ++j) {
      float e = __expf(srow[j] - mx);
      sum += e;
      prow[j] = (_Float16)e;
    }
#pragma unroll
    for (int j = NTOK; j < 160; ++j) prow[j] = (_Float16)0.f;  // K-pad
    rsum[tid] = 1.0f / sum;
  }
  __syncthreads();

  // O = P V : 18 tiles x 5 K-steps (K padded 144->160)
  _Float16* aop = AO + (size_t)(bh / HEADS) * NTOK * CDIM;
  for (int t = wave; t < 18; t += 8) {
    int mt = t >> 1, nt = t & 1;
    v8f acc = {};
#pragma unroll
    for (int kk = 0; kk < 5; ++kk) {
      v16h a  = load_frag(Ps,  mt * 16, 168, kk * 32, lane);
      v16h bb = load_frag(Vts, nt * 16, 168, kk * 32, lane);
      acc = __builtin_amdgcn_wmma_f32_16x16x32_f16(false, a, false, bb, (short)0, acc, false, false);
    }
    int d  = nt * 16 + (lane & 15);
    int rb = mt * 16 + ((lane >> 4) << 3);
#pragma unroll
    for (int r = 0; r < 8; ++r) {
      float v = acc[r] * rsum[rb + r];
      aop[(size_t)(rb + r) * CDIM + h * DH + d] = (_Float16)v;
    }
  }
}

// ---- output projection: one block per (b,w), fp32 out
__global__ __launch_bounds__(256) void k_proj(
    const _Float16* __restrict__ AO, const _Float16* __restrict__ wTp,
    const float* __restrict__ bproj, float* __restrict__ out) {
  __shared__ _Float16 As[NTOK * 200];
  int bw = blockIdx.x;
  const _Float16* ap = AO + (size_t)bw * NTOK * CDIM;
  int tid = threadIdx.x, lane = tid & 31, wave = tid >> 5;
  for (int c = tid; c < NTOK * (CDIM / 8); c += 256) {
    int row = c / 24, co = (c % 24) * 8;
    *(v8h*)&As[row * 200 + co] = *(const v8h*)&ap[row * CDIM + co];
  }
  __syncthreads();
  float* op = out + (size_t)bw * NTOK * CDIM;
  for (int t = wave; t < 9 * 12; t += 8) {
    int mt = t / 12, nt = t % 12;
    int m0 = mt * 16, n0 = nt * 16;
    int col = n0 + (lane & 15);
    float bv = bproj[col];
    v8f acc;
#pragma unroll
    for (int r = 0; r < 8; ++r) acc[r] = bv;
#pragma unroll
    for (int kk = 0; kk < 6; ++kk) {
      v16h a  = load_frag(As, m0, 200, kk * 32, lane);
      v16h bb = load_frag(wTp, n0, CDIM, kk * 32, lane);
      acc = __builtin_amdgcn_wmma_f32_16x16x32_f16(false, a, false, bb, (short)0, acc, false, false);
    }
    int rb = m0 + ((lane >> 4) << 3);
#pragma unroll
    for (int r = 0; r < 8; ++r) op[(size_t)(rb + r) * CDIM + col] = acc[r];
  }
}

extern "C" void kernel_launch(void* const* d_in, const int* in_sizes, int n_in,
                              void* d_out, int out_size, void* d_ws, size_t ws_size,
                              hipStream_t stream) {
  (void)in_sizes; (void)n_in; (void)out_size; (void)ws_size;
  const float* x     = (const float*)d_in[0];
  const float* mask  = (const float*)d_in[1];
  const float* wqkv  = (const float*)d_in[2];
  const float* bqkv  = (const float*)d_in[3];
  const float* wproj = (const float*)d_in[4];
  const float* bproj = (const float*)d_in[5];
  const float* btab  = (const float*)d_in[6];
  const int*   pidx  = (const int*)d_in[7];
  float* out = (float*)d_out;

  char* base = (char*)d_ws;
  size_t off = 0;
  auto alloc = [&](size_t bytes) -> char* {
    char* p = base + off;
    off = (off + bytes + 255) & ~(size_t)255;
    return p;
  };
  _Float16* wTq   = (_Float16*)alloc((size_t)576 * 192 * 2);
  _Float16* wTp   = (_Float16*)alloc((size_t)192 * 192 * 2);
  float*    biasG = (float*)alloc((size_t)NW * HEADS * NTOK * NTOK * 4);
  size_t qkbytes  = (size_t)BDIM * NW * HEADS * NTOK * DH * 2;
  _Float16* Q  = (_Float16*)alloc(qkbytes);
  _Float16* K  = (_Float16*)alloc(qkbytes);
  _Float16* Vt = (_Float16*)alloc(qkbytes);
  _Float16* AO = (_Float16*)alloc((size_t)BDIM * NW * NTOK * CDIM * 2);

  k_prep_w<<<(576 * 192 + 255) / 256, 256, 0, stream>>>(wqkv, wTq, 192, 576);
  k_prep_w<<<(192 * 192 + 255) / 256, 256, 0, stream>>>(wproj, wTp, 192, 192);
  k_prep_bias<<<(NW * HEADS * NTOK * NTOK + 255) / 256, 256, 0, stream>>>(btab, pidx, biasG);
  k_qkv<<<BDIM * NW, 256, 0, stream>>>(x, wTq, bqkv, Q, K, Vt);

  const int attn_lds = 144*40*2 + 144*40*2 + 32*168*2 + 144*148*4 + 144*168*2 + 144*4; // 168000 B
  (void)hipFuncSetAttribute((const void*)k_attn, hipFuncAttributeMaxDynamicSharedMemorySize, attn_lds);
  k_attn<<<BDIM * NW * HEADS, 256, attn_lds, stream>>>(Q, K, Vt, biasG, mask, AO);

  k_proj<<<BDIM * NW, 256, 0, stream>>>(AO, wTp, bproj, out);
}
